// GroupedQueryAttention_47347719471595
// MI455X (gfx1250) — matrix operand order repaired and checked by reading, not compiled, not run on hardware
//
#include <hip/hip_runtime.h>

#define BB   2
#define SS   2048
#define DM   2048
#define NH   32
#define NKV  8
#define HD   64
#define NREP (NH / NKV)

typedef __attribute__((ext_vector_type(16))) __bf16 v16bf;
typedef __attribute__((ext_vector_type(8)))  __bf16 v8bf;
typedef __attribute__((ext_vector_type(8)))  float  v8f;

// ---------- WMMA operand loaders (contiguous k = 16*half + s, same order for A and B) ----------
// A 16x32 (MxK): lane L(<16): row M=L, K=0..15; lane L+16: row M=L, K=16..31
// caller passes p = rowBase + (half*16); we fetch 16 contiguous K values.
__device__ __forceinline__ v16bf ldA16(const __bf16* p) {
  v8bf lo = *(const v8bf*)(p);
  v8bf hi = *(const v8bf*)(p + 8);
  return __builtin_shufflevector(lo, hi, 0,1,2,3,4,5,6,7,8,9,10,11,12,13,14,15);
}
// B 32x16 (KxN): lane L(<16): col N=L, K=0..15; lane L+16: col N=L, K=16..31
// caller passes p = colBase + (half*16); we fetch 16 contiguous K values.
__device__ __forceinline__ v16bf ldB16(const __bf16* p) {
  v8bf lo = *(const v8bf*)(p);
  v8bf hi = *(const v8bf*)(p + 8);
  return __builtin_shufflevector(lo, hi, 0,1,2,3,4,5,6,7,8,9,10,11,12,13,14,15);
}
__device__ __forceinline__ v8f wmma_bf16(v16bf a, v16bf b, v8f c) {
  return __builtin_amdgcn_wmma_f32_16x16x32_bf16(false, a, false, b, (short)0, c, false, false);
}

// ---------- f32 -> bf16 convert ----------
__global__ __launch_bounds__(256) void cvt_bf16(const float* __restrict__ src,
                                                __bf16* __restrict__ dst, int n) {
  int i = blockIdx.x * 256 + threadIdx.x;
  if (i < n) dst[i] = (__bf16)src[i];
}

// ---------- blocked GEMM: C[M,N](f32) = A[M,K](bf16) @ B[N,K](bf16)^T ----------
// 256 threads = 8 waves per block; block computes a 128x128 tile of C.
// A/B 128x32 panels staged in LDS with GLOBAL_LOAD_ASYNC_TO_LDS (ASYNCcnt),
// double-buffered. Wave w -> 64x32 sub-tile: rows (w&1)*64.., cols (w>>1)*32..
__global__ __launch_bounds__(256) void gemm_bf16_nt(const __bf16* __restrict__ A,
                                                    const __bf16* __restrict__ Bw,
                                                    float* __restrict__ C,
                                                    int M, int N, int K) {
  __shared__ __attribute__((aligned(16))) __bf16 sA[2][128 * 32];  // 2 x 8 KB
  __shared__ __attribute__((aligned(16))) __bf16 sB[2][128 * 32];  // 2 x 8 KB

  const int nTileN = N >> 7;
  const int bm = blockIdx.x / nTileN, bn = blockIdx.x % nTileN;
  const int tm0 = bm << 7, tn0 = bn << 7;

  const int t    = threadIdx.x;
  const int lane = t & 31, half = lane >> 4, lm = lane & 15;
  const int w    = t >> 5;
  const int wr   = w & 1;        // row half of the 128-row tile (64 rows)
  const int wc   = w >> 1;       // 32-col strip (4 strips)

  // cooperative staging map: thread t moves 32 bytes: row (t>>1), 32B segment (t&1)
  const int lr   = t >> 1;                 // 0..127
  const int lsg  = (t & 1) * 16;           // element offset in row (16 bf16 = 32 B)
  const __bf16* ga = A  + (size_t)(tm0 + lr) * K + lsg;
  const __bf16* gb = Bw + (size_t)(tn0 + lr) * K + lsg;
  const uint32_t ldsA0 = (uint32_t)(uintptr_t)&sA[0][lr * 32 + lsg];
  const uint32_t ldsB0 = (uint32_t)(uintptr_t)&sB[0][lr * 32 + lsg];

#define STAGE(bufsel, gaa, gbb)                                                              \
  do {                                                                                       \
    uint32_t la_ = ldsA0 + (uint32_t)(bufsel) * (128 * 32 * 2);                              \
    uint32_t lb_ = ldsB0 + (uint32_t)(bufsel) * (128 * 32 * 2);                              \
    asm volatile("global_load_async_to_lds_b128 %0, %1, off"           :: "v"(la_), "v"(gaa) : "memory"); \
    asm volatile("global_load_async_to_lds_b128 %0, %1, off offset:16" :: "v"(la_), "v"(gaa) : "memory"); \
    asm volatile("global_load_async_to_lds_b128 %0, %1, off"           :: "v"(lb_), "v"(gbb) : "memory"); \
    asm volatile("global_load_async_to_lds_b128 %0, %1, off offset:16" :: "v"(lb_), "v"(gbb) : "memory"); \
  } while (0)

  v8f acc[4][2] = {};
  const int nk = K >> 5;

  STAGE(0, ga, gb);                       // prologue: fill buffer 0
  for (int i = 0; i < nk; ++i) {
    const int cur = i & 1;
    if (i + 1 < nk) {
      ga += 32; gb += 32;
      STAGE(cur ^ 1, ga, gb);             // prefetch next panel (4 async instrs/wave)
      asm volatile("s_wait_asynccnt 4" ::: "memory");   // current panel complete
    } else {
      asm volatile("s_wait_asynccnt 0" ::: "memory");
    }
    __syncthreads();                      // panel visible to all waves

    v16bf af[4], bfr[2];
#pragma unroll
    for (int m = 0; m < 4; ++m)
      af[m] = ldA16(&sA[cur][(wr * 64 + m * 16 + lm) * 32 + half * 16]);
#pragma unroll
    for (int n = 0; n < 2; ++n)
      bfr[n] = ldB16(&sB[cur][(wc * 32 + n * 16 + lm) * 32 + half * 16]);
#pragma unroll
    for (int m = 0; m < 4; ++m)
#pragma unroll
      for (int n = 0; n < 2; ++n)
        acc[m][n] = wmma_bf16(af[m], bfr[n], acc[m][n]);

    asm volatile("s_wait_dscnt 0" ::: "memory");        // LDS reads done
    __syncthreads();                      // safe to overwrite this buffer
  }
#undef STAGE

  // C/D layout: VGPR r -> row (r + half*8), col = lm
#pragma unroll
  for (int m = 0; m < 4; ++m)
#pragma unroll
    for (int n = 0; n < 2; ++n) {
      float* crow = C + (size_t)(tm0 + wr * 64 + m * 16 + half * 8) * N
                      + tn0 + wc * 32 + n * 16 + lm;
#pragma unroll
      for (int r = 0; r < 8; ++r) crow[(size_t)r * N] = acc[m][n][r];
    }
}

// ---------- RoPE (folds scale for Q), write bf16 [B,nheads,S,HD] ----------
__global__ __launch_bounds__(256) void rope_q_k(const float* __restrict__ pre,
                                                __bf16* __restrict__ outb,
                                                int nheads, float scale) {
  size_t i = (size_t)blockIdx.x * 256 + threadIdx.x;   // over B*S*nheads*HD
  int d = (int)(i & 63);
  size_t t = i >> 6;
  int h = (int)(t % nheads);
  size_t bs = t / nheads;                               // b*S + s
  int s = (int)(bs & (SS - 1));
  int b = (int)(bs >> 11);

  float x = pre[i];
  int dm = d & 31;
  float inv = __expf(-9.210340371976184f * ((float)dm * (1.0f / 32.0f))); // 10000^(-dm/32)
  float ang = (float)s * inv;
  float sn, cs;
  __sincosf(ang, &sn, &cs);
  float partner = (d < 32) ? -pre[i + 32] : pre[i - 32];
  float o = (x * cs + partner * sn) * scale;
  outb[(((size_t)(b * nheads + h)) * SS + s) * HD + d] = (__bf16)o;
}

// ---------- V transpose: f32 [B,S,KV*HD] -> bf16 [B,KV,HD,S] ----------
__global__ __launch_bounds__(256) void v_transpose(const float* __restrict__ vp,
                                                   __bf16* __restrict__ vT) {
  size_t i = (size_t)blockIdx.x * 256 + threadIdx.x;   // over B*S*NKV*HD
  int d = (int)(i & 63);
  size_t t = i >> 6;
  int h = (int)(t & (NKV - 1));
  size_t bs = t >> 3;
  int s = (int)(bs & (SS - 1));
  int b = (int)(bs >> 11);
  vT[(((size_t)(b * NKV + h)) * HD + d) * SS + s] = (__bf16)vp[i];
}

// ---------- flash attention: one wave per (b,h,16-row Q tile) ----------
__global__ __launch_bounds__(128) void attn_fwd(const __bf16* __restrict__ Q,   // [B,NH,S,HD]
                                                const __bf16* __restrict__ Kk,  // [B,NKV,S,HD]
                                                const __bf16* __restrict__ Vt,  // [B,NKV,HD,S]
                                                __bf16* __restrict__ Oa) {      // [B,S,NH*HD]
  const int WAVES = 4;
  __shared__ __attribute__((aligned(16))) float  s_sc[WAVES][16 * 32];
  __shared__ __attribute__((aligned(16))) __bf16 s_p [WAVES][16 * 32];
  __shared__ float s_alpha[WAVES][16];
  __shared__ float s_l[WAVES][16];

  int w    = threadIdx.x >> 5;
  int wave = blockIdx.x * WAVES + w;
  const int QT = SS / 16;
  int qt  = wave % QT;
  int h   = (wave / QT) % NH;
  int b   = wave / (QT * NH);
  int kvh = h / NREP;
  int lane = threadIdx.x & 31, half = lane >> 4, lm = lane & 15;
  int q0 = qt * 16;

  const __bf16* qbase = Q  + (((size_t)(b * NH + h)) * SS + q0 + lm) * HD + half * 16;
  const __bf16* kbase = Kk + ((size_t)(b * NKV + kvh)) * SS * HD;
  const __bf16* vbase = Vt + ((size_t)(b * NKV + kvh)) * HD * SS;

  v16bf aq0 = ldA16(qbase);        // head-dim K chunk 0..31
  v16bf aq1 = ldA16(qbase + 32);   // head-dim K chunk 32..63

  v8f o0 = {}, o1 = {}, o2 = {}, o3 = {};
  float mrow = -__builtin_inff(), lrow = 0.f;   // valid in lanes 0..15 (lane == local q row)
  int   qg   = q0 + lm;                         // global q position for this lane's row
  int   kend = q0 + 16;                         // causal: keys 0 .. q0+15

  for (int kb = 0; kb < kend; kb += 32) {
    // ---- scores S = Q @ K^T  (two 16x16 tiles: keys kb..kb+15, kb+16..kb+31) ----
    const __bf16* b0 = kbase + (size_t)(kb + lm) * HD + half * 16;
    const __bf16* b1 = kbase + (size_t)(kb + 16 + lm) * HD + half * 16;
    v8f c0 = {}, c1 = {};
    c0 = wmma_bf16(aq0, ldB16(b0), c0);
    c0 = wmma_bf16(aq1, ldB16(b0 + 32), c0);
    c1 = wmma_bf16(aq0, ldB16(b1), c1);
    c1 = wmma_bf16(aq1, ldB16(b1 + 32), c1);

    int row = half * 8;
    for (int r = 0; r < 8; ++r) {
      s_sc[w][(row + r) * 32 + lm]      = c0[r];
      s_sc[w][(row + r) * 32 + 16 + lm] = c1[r];
    }
    asm volatile("s_wait_dscnt 0" ::: "memory");

    // ---- online softmax: lane r (<16) owns q row r of the tile ----
    if (lane < 16) {
      float mnew = mrow;
      for (int j = 0; j < 32; ++j) {
        float s = s_sc[w][lane * 32 + j];
        if (kb + j > qg) s -= 1.0e9f;           // causal mask (== reference triu -1e9)
        mnew = fmaxf(mnew, s);
      }
      float alpha = __expf(mrow - mnew);
      float ls = 0.f;
      for (int j = 0; j < 32; ++j) {
        float s = s_sc[w][lane * 32 + j];
        if (kb + j > qg) s -= 1.0e9f;
        float p = __expf(s - mnew);
        ls += p;
        s_p[w][lane * 32 + j] = (__bf16)p;
      }
      mrow = mnew;
      lrow = lrow * alpha + ls;
      s_alpha[w][lane] = alpha;
    }
    asm volatile("s_wait_dscnt 0" ::: "memory");

    // ---- rescale running O by alpha of each row ----
    for (int r = 0; r < 8; ++r) {
      float al = s_alpha[w][half * 8 + r];
      o0[r] *= al; o1[r] *= al; o2[r] *= al; o3[r] *= al;
    }

    // ---- O += P(16x32) @ V(32x64): P as A-operand from LDS, V^T columns contiguous ----
    v16bf ap = ldA16(&s_p[w][lm * 32 + half * 16]);
    const __bf16* vb = vbase + (size_t)lm * SS + kb + half * 16;
    o0 = wmma_bf16(ap, ldB16(vb + (size_t)0 * 16 * SS), o0);
    o1 = wmma_bf16(ap, ldB16(vb + (size_t)1 * 16 * SS), o1);
    o2 = wmma_bf16(ap, ldB16(vb + (size_t)2 * 16 * SS), o2);
    o3 = wmma_bf16(ap, ldB16(vb + (size_t)3 * 16 * SS), o3);
  }

  if (lane < 16) s_l[w][lane] = lrow;
  asm volatile("s_wait_dscnt 0" ::: "memory");

  // ---- normalize and write bf16 [B,S, h*64 + d] for the O-projection GEMM ----
  __bf16* obase = Oa + ((size_t)b * SS + q0 + half * 8) * DM + h * HD + lm;
  for (int r = 0; r < 8; ++r) {
    float linv = 1.0f / s_l[w][half * 8 + r];
    size_t off = (size_t)r * DM;
    obase[off + 0]  = (__bf16)(o0[r] * linv);
    obase[off + 16] = (__bf16)(o1[r] * linv);
    obase[off + 32] = (__bf16)(o2[r] * linv);
    obase[off + 48] = (__bf16)(o3[r] * linv);
  }
}

extern "C" void kernel_launch(void* const* d_in, const int* in_sizes, int n_in,
                              void* d_out, int out_size, void* d_ws, size_t ws_size,
                              hipStream_t stream) {
  const float* x  = (const float*)d_in[0];
  const float* Wq = (const float*)d_in[1];
  const float* Wk = (const float*)d_in[2];
  const float* Wv = (const float*)d_in[3];
  const float* Wo = (const float*)d_in[4];
  // d_in[5] = mask: causal -1e9 triu, computed analytically in attn_fwd.
  float* out = (float*)d_out;

  const size_t nX  = (size_t)BB * SS * DM;        // 8,388,608
  const size_t nWq = (size_t)DM * DM;             // 4,194,304
  const size_t nWk = (size_t)(NKV * HD) * DM;     // 1,048,576
  const size_t nKV = (size_t)BB * SS * NKV * HD;  // 2,097,152

  char* p = (char*)d_ws;
  auto carve = [&](size_t bytes) { char* r = p; p += (bytes + 255) & ~(size_t)255; return r; };
  __bf16* xb    = (__bf16*)carve(nX  * 2);
  __bf16* Wqb   = (__bf16*)carve(nWq * 2);
  __bf16* Wkb   = (__bf16*)carve(nWk * 2);
  __bf16* Wvb   = (__bf16*)carve(nWk * 2);
  __bf16* Wob   = (__bf16*)carve(nWq * 2);
  float*  q_pre = (float*) carve(nX  * 4);
  float*  k_pre = (float*) carve(nKV * 4);
  float*  v_pre = (float*) carve(nKV * 4);
  __bf16* qb    = (__bf16*)carve(nX  * 2);
  __bf16* kbf   = (__bf16*)carve(nKV * 2);
  __bf16* vT    = (__bf16*)carve(nKV * 2);
  __bf16* attn  = xb;  // xb is dead after the QKV GEMMs; same size -> alias

  // 1) convert inputs to bf16
  cvt_bf16<<<(int)(nX  / 256), 256, 0, stream>>>(x,  xb,  (int)nX);
  cvt_bf16<<<(int)(nWq / 256), 256, 0, stream>>>(Wq, Wqb, (int)nWq);
  cvt_bf16<<<(int)(nWk / 256), 256, 0, stream>>>(Wk, Wkb, (int)nWk);
  cvt_bf16<<<(int)(nWk / 256), 256, 0, stream>>>(Wv, Wvb, (int)nWk);
  cvt_bf16<<<(int)(nWq / 256), 256, 0, stream>>>(Wo, Wob, (int)nWq);

  // 2) QKV projections: [4096, K=2048] @ W^T  (128x128 tile per 256-thread block)
  const int M = BB * SS;
  gemm_bf16_nt<<<(M / 128) * (DM / 128), 256, 0, stream>>>(xb, Wqb, q_pre, M, DM, DM);
  gemm_bf16_nt<<<(M / 128) * ((NKV * HD) / 128), 256, 0, stream>>>(xb, Wkb, k_pre, M, NKV * HD, DM);
  gemm_bf16_nt<<<(M / 128) * ((NKV * HD) / 128), 256, 0, stream>>>(xb, Wvb, v_pre, M, NKV * HD, DM);

  // 3) RoPE (+1/sqrt(64) folded into Q), V transpose
  rope_q_k<<<(int)(nX  / 256), 256, 0, stream>>>(q_pre, qb,  NH,  0.125f);
  rope_q_k<<<(int)(nKV / 256), 256, 0, stream>>>(k_pre, kbf, NKV, 1.0f);
  v_transpose<<<(int)(nKV / 256), 256, 0, stream>>>(v_pre, vT);

  // 4) causal flash attention (one wave per 16-row q tile): B*NH*(S/16) = 8192 waves
  attn_fwd<<<(BB * NH * (SS / 16)) / 4, 128, 0, stream>>>(qb, kbf, vT, attn);

  // 5) output projection -> f32 d_out
  gemm_bf16_nt<<<(M / 128) * (DM / 128), 256, 0, stream>>>(attn, Wob, out, M, DM, DM);
}
